// GCN_3161095930269
// MI455X (gfx1250) — matrix operand order repaired and checked by reading, not compiled, not run on hardware
//
#include <hip/hip_runtime.h>

// ---------------------------------------------------------------------------
// Fused GCN forward for MI455X (gfx1250).
// One workgroup (8 wave32) per batch element; all intermediates in LDS (bf16);
// matrix math on v_wmma_f32_16x16x32_bf16. Hidden dim processed in 8 chunks of
// 32 columns so total LDS (~156 KB) fits 2 workgroups/WGP for load/compute
// overlap across blocks (workload is HBM-bound: ~440 MB @ 23.3 TB/s ~= 19 us).
// ---------------------------------------------------------------------------

typedef __attribute__((ext_vector_type(16))) __bf16        v16bf;
typedef __attribute__((ext_vector_type(2)))  __bf16        v2bf;
typedef __attribute__((ext_vector_type(8)))  float         v8f;
typedef __attribute__((ext_vector_type(2)))  float         v2f;
typedef __attribute__((ext_vector_type(4)))  float         v4f;
typedef __attribute__((ext_vector_type(8)))  unsigned int  v8u;
typedef __attribute__((ext_vector_type(4)))  unsigned int  v4u;

#define B_SZ    4096
#define N_NODES 116
#define NP      128      // padded node/feature dim
#define HID     256
#define WAVES   8
#define CHUNK   32       // hidden columns per chunk
#define NCHUNK  (HID / CHUNK)

#define SXP  136         // padded K-stride for X/S/W1Tc/XWTc/XW2T (avoids 64-bank aliasing)
#define SH1  264         // padded K-stride for H1/W2T (K = 256)

// ---- LDS layout (bytes) ----------------------------------------------------
#define OFF_S     0u        // bf16 [128][SXP]   support (node-major, K contiguous)
#define OFF_X     34816u    // bf16 [128][SXP]   x
#define OFF_H1    69632u    // bf16 [128][SH1]   H1 (node-major, K=hid contiguous)
#define OFF_CH    137216u   // chunk region (17408 B), aliased by tail phases:
#define OFF_W1TC  137216u   //   bf16 [32][SXP]  W1 chunk transposed ([hid'][feat])
#define OFF_XWTC  145920u   //   bf16 [32][SXP]  XW chunk transposed ([hid'][node])
#define OFF_W2T   137216u   //   bf16 [16][SH1]  W2 transposed (alias, phases 3+)
#define OFF_XW2T  145920u   //   bf16 [16][SXP]  XW2 transposed (alias)
#define OFF_H2    150272u   //   f32  [128][4]   (alias)
#define OFF_B1    154624u   // f32 [256]
#define OFF_B2    155648u   // f32 [4] (padded)
#define OFF_R     155712u   // f32 [64]
#define LDS_TOTAL 155968u   // < 160 KB  =>  2 workgroups per WGP

// float -> bf16 via native convert (v_cvt_*bf16* on gfx1250, RNE)
static __device__ inline unsigned pack_bf2(float a, float b) {
  v2f f = {a, b};
  v2bf h = __builtin_convertvector(f, v2bf);
  return __builtin_bit_cast(unsigned, h);
}
static __device__ inline unsigned short f2bf(float f) {
  __bf16 h = (__bf16)f;
  return __builtin_bit_cast(unsigned short, h);
}

// A fragment (16x32 bf16), operand row-major with K contiguous.
// ISA layout: lanes 0-15 hold K{0..7,16..23}, lanes 16-31 hold K{8..15,24..31}.
static __device__ inline v16bf load_frag_A(const unsigned short* base, int rowBase,
                                           int strideElems, int kBase, int lane) {
  const int m  = lane & 15;
  const int hi = lane >> 4;
  const v4u* q = (const v4u*)(base + (rowBase + m) * strideElems + kBase + 8 * hi);
  const v4u lo = q[0];  // K = 0..7   (+8*hi)
  const v4u up = q[2];  // K = 16..23 (+8*hi)
  v8u w;
  w[0] = lo[0]; w[1] = lo[1]; w[2] = lo[2]; w[3] = lo[3];
  w[4] = up[0]; w[5] = up[1]; w[6] = up[2]; w[7] = up[3];
  return __builtin_bit_cast(v16bf, w);
}

// B fragment (32x16 bf16), operand stored TRANSPOSED ([col][k], K contiguous).
// ISA layout: lanes 0-15 hold K{0..7,16..23}, lanes 16-31 hold K{8..15,24..31} for column n.
static __device__ inline v16bf load_frag_BT(const unsigned short* base, int colBase,
                                            int strideElems, int kBase, int lane) {
  const int n  = lane & 15;
  const int hi = lane >> 4;
  const v4u* q = (const v4u*)(base + (colBase + n) * strideElems + kBase + 8 * hi);
  const v4u a = q[0];
  const v4u b = q[2];
  v8u w;
  w[0] = a[0]; w[1] = a[1]; w[2] = a[2]; w[3] = a[3];
  w[4] = b[0]; w[5] = b[1]; w[6] = b[2]; w[7] = b[3];
  return __builtin_bit_cast(v16bf, w);
}

static __device__ inline v8f wmma_bf16(v16bf a, v16bf b, v8f c) {
  return __builtin_amdgcn_wmma_f32_16x16x32_bf16(false, a, false, b, (short)0, c,
                                                 false, false);
}

__global__ __launch_bounds__(256)
void gcn_fused_kernel(const float* __restrict__ x,   const float* __restrict__ support,
                      const float* __restrict__ W1,  const float* __restrict__ b1,
                      const float* __restrict__ W2,  const float* __restrict__ b2,
                      const float* __restrict__ Wr1, const float* __restrict__ br1,
                      const float* __restrict__ Wr2, const float* __restrict__ br2,
                      float* __restrict__ out) {
  extern __shared__ char smem[];
  unsigned short* sS    = (unsigned short*)(smem + OFF_S);
  unsigned short* sX    = (unsigned short*)(smem + OFF_X);
  unsigned short* sH1   = (unsigned short*)(smem + OFF_H1);
  unsigned short* sW1TC = (unsigned short*)(smem + OFF_W1TC);
  unsigned short* sXWTC = (unsigned short*)(smem + OFF_XWTC);
  unsigned short* sW2T  = (unsigned short*)(smem + OFF_W2T);
  unsigned short* sXW2T = (unsigned short*)(smem + OFF_XW2T);
  float*          sH2   = (float*)(smem + OFF_H2);
  float*          sB1   = (float*)(smem + OFF_B1);
  float*          sB2   = (float*)(smem + OFF_B2);
  float*          sR    = (float*)(smem + OFF_R);

  const int b    = blockIdx.x;
  const int tid  = threadIdx.x;
  const int lane = tid & 31;
  const int wave = tid >> 5;
  const int hi   = lane >> 4;
  const int ln   = lane & 15;

  // -------- Phase 0a: zero padded regions (X, S, chunk region), stage biases -
  {
    unsigned* z = (unsigned*)(smem + OFF_S);          // S + X contiguous
    for (int i = tid; i < (int)((34816u * 2) / 4); i += 256) z[i] = 0u;
    unsigned* zc = (unsigned*)(smem + OFF_CH);        // W1TC + XWTC
    for (int i = tid; i < (int)(17408u / 4); i += 256) zc[i] = 0u;
  }
  if (tid < HID) sB1[tid] = b1[tid];
  if (tid < 4)   sB2[tid] = b2[tid];
  __syncthreads();

  // -------- Phase 0b: stage x, support as bf16 (coalesced float4 loads) ------
  // 116*116 floats per matrix; 116 % 4 == 0 so float4 never crosses a row.
  {
    const v4f* xb4 = (const v4f*)(x       + (size_t)b * N_NODES * N_NODES);
    const v4f* sb4 = (const v4f*)(support + (size_t)b * N_NODES * N_NODES);
    for (int i4 = tid; i4 < (N_NODES * N_NODES) / 4; i4 += 256) {
      const int r  = i4 / 29;            // 116/4 = 29 float4 per row
      const int c4 = (i4 - r * 29) * 4;
      const v4f xv = xb4[i4];
      const v4f sv = sb4[i4];
      unsigned long long xp = (unsigned long long)pack_bf2(xv[0], xv[1]) |
                              ((unsigned long long)pack_bf2(xv[2], xv[3]) << 32);
      unsigned long long sp = (unsigned long long)pack_bf2(sv[0], sv[1]) |
                              ((unsigned long long)pack_bf2(sv[2], sv[3]) << 32);
      *(unsigned long long*)(sX + r * SXP + c4) = xp;
      *(unsigned long long*)(sS + r * SXP + c4) = sp;
    }
  }

  // -------- Chunked GraphConv1: for each 32-wide hidden chunk ---------------
  for (int c = 0; c < NCHUNK; ++c) {
    // Stage W1 chunk transposed: sW1TC[h'][k] = W1[k][c*32+h'].
    // (Overlaps previous chunk's phase-2 compute; region is disjoint.)
    for (int idx = tid; idx < N_NODES * CHUNK; idx += 256) {
      const int k = idx >> 5, h = idx & 31;
      sW1TC[h * SXP + k] = f2bf(W1[k * HID + c * CHUNK + h]);
    }
    __syncthreads();  // staging done AND previous chunk's phase 2 done

    // Phase 1: XWc = X @ W1c  -> sXWTC[h'][node]
#pragma unroll
    for (int tt = 0; tt < 2; ++tt) {
      const int mt  = wave * 16;
      const int ntc = tt * 16;
      v8f acc = {};
#pragma unroll
      for (int kb = 0; kb < 4; ++kb) {
        v16bf a  = load_frag_A (sX,    mt,  SXP, kb * 32, lane);
        v16bf bb = load_frag_BT(sW1TC, ntc, SXP, kb * 32, lane);
        acc = wmma_bf16(a, bb, acc);
      }
      v4u d;
#pragma unroll
      for (int p = 0; p < 4; ++p) d[p] = pack_bf2(acc[2 * p], acc[2 * p + 1]);
      *(v4u*)(sXWTC + (ntc + ln) * SXP + mt + 8 * hi) = d;
    }
    __syncthreads();

    // Phase 2: H1c = relu(S @ XWc + b1c) -> sH1[node][c*32 + h']
#pragma unroll
    for (int tt = 0; tt < 2; ++tt) {
      const int mt  = wave * 16;
      const int ntc = tt * 16;
      v8f acc = {};
#pragma unroll
      for (int kb = 0; kb < 4; ++kb) {
        v16bf a  = load_frag_A (sS,    mt,  SXP, kb * 32, lane);
        v16bf bb = load_frag_BT(sXWTC, ntc, SXP, kb * 32, lane);
        acc = wmma_bf16(a, bb, acc);
      }
      const int n = c * CHUNK + ntc + ln;
      const float bias = sB1[n];
#pragma unroll
      for (int r = 0; r < 8; ++r) {
        float v = acc[r] + bias;
        v = v > 0.f ? v : 0.f;
        sH1[(mt + r + 8 * hi) * SH1 + n] = f2bf(v);
      }
    }
    // no barrier: next chunk's staging touches only sW1TC (not read here)
  }
  __syncthreads();  // H1 complete; chunk region free for reuse

  // -------- Stage W2 transposed (out padded 4 -> 16) ------------------------
  for (int idx = tid; idx < 16 * HID; idx += 256) {
    const int n = idx >> 8, k = idx & 255;
    const float v = (n < 4) ? W2[k * 4 + n] : 0.f;
    sW2T[n * SH1 + k] = f2bf(v);
  }
  __syncthreads();

  // -------- Phase 3: XW2 = H1 @ W2 -> sXW2T[n][node] ------------------------
  {
    const int mt = wave * 16;
    v8f acc = {};
#pragma unroll
    for (int kb = 0; kb < 8; ++kb) {
      v16bf a  = load_frag_A (sH1,  mt, SH1, kb * 32, lane);
      v16bf bb = load_frag_BT(sW2T, 0,  SH1, kb * 32, lane);
      acc = wmma_bf16(a, bb, acc);
    }
    v4u d;
#pragma unroll
    for (int p = 0; p < 4; ++p) d[p] = pack_bf2(acc[2 * p], acc[2 * p + 1]);
    *(v4u*)(sXW2T + ln * SXP + mt + 8 * hi) = d;
  }
  __syncthreads();

  // -------- Phase 4: H2 = relu(S @ XW2 + b2) -> sH2[node][4] (f32) ----------
  {
    const int mt = wave * 16;
    v8f acc = {};
#pragma unroll
    for (int kb = 0; kb < 4; ++kb) {
      v16bf a  = load_frag_A (sS,    mt, SXP, kb * 32, lane);
      v16bf bb = load_frag_BT(sXW2T, 0,  SXP, kb * 32, lane);
      acc = wmma_bf16(a, bb, acc);
    }
    if (ln < 4) {
      const float bias = sB2[ln];
#pragma unroll
      for (int r = 0; r < 8; ++r) {
        const int m = mt + r + 8 * hi;
        if (m < N_NODES) {
          float v = acc[r] + bias;
          sH2[m * 4 + ln] = v > 0.f ? v : 0.f;
        }
      }
    }
  }
  __syncthreads();

  // -------- Phase 5: r = relu(flat @ Wr1 + br1), flat = sH2[0..463] ---------
  if (tid < 64) {
    float sum = br1[tid];
    for (int i = 0; i < N_NODES * 4; ++i)
      sum += sH2[i] * Wr1[i * 64 + tid];
    sR[tid] = sum > 0.f ? sum : 0.f;
  }
  __syncthreads();

  // -------- Phase 6: logits + log_softmax -----------------------------------
  if (tid == 0) {
    float l0 = br2[0], l1 = br2[1];
    for (int j = 0; j < 64; ++j) {
      const float rv = sR[j];
      l0 += rv * Wr2[j * 2 + 0];
      l1 += rv * Wr2[j * 2 + 1];
    }
    const float mx  = l0 > l1 ? l0 : l1;
    const float lse = mx + __logf(__expf(l0 - mx) + __expf(l1 - mx));
    out[b * 2 + 0] = l0 - lse;
    out[b * 2 + 1] = l1 - lse;
  }
}

extern "C" void kernel_launch(void* const* d_in, const int* in_sizes, int n_in,
                              void* d_out, int out_size, void* d_ws, size_t ws_size,
                              hipStream_t stream) {
  (void)in_sizes; (void)n_in; (void)out_size; (void)d_ws; (void)ws_size;
  const float* x       = (const float*)d_in[0];
  const float* support = (const float*)d_in[1];
  const float* W1      = (const float*)d_in[2];
  const float* b1      = (const float*)d_in[3];
  const float* W2      = (const float*)d_in[4];
  const float* b2      = (const float*)d_in[5];
  const float* Wr1     = (const float*)d_in[6];
  const float* br1     = (const float*)d_in[7];
  const float* Wr2     = (const float*)d_in[8];
  const float* br2     = (const float*)d_in[9];
  float* out = (float*)d_out;

  gcn_fused_kernel<<<dim3(B_SZ), dim3(256), LDS_TOTAL, stream>>>(
      x, support, W1, b1, W2, b2, Wr1, br1, Wr2, br2, out);
}
